// MambaBlock_59700045414691
// MI455X (gfx1250) — compile-verified, matrix operand order checked by reading, not run on hardware
//
#include <hip/hip_runtime.h>
#include <hip/hip_bf16.h>
#include <math.h>

typedef __bf16 bf16;
typedef __attribute__((ext_vector_type(16))) __bf16        v16bf;
typedef __attribute__((ext_vector_type(8)))  float         v8f;
typedef __attribute__((ext_vector_type(4)))  unsigned int  v4u;

#define TOK   2048   // B * L tokens
#define LSEQ  1024
#define NB    2
#define DM    256    // d_model
#define DI    512    // d_inner
#define DS    16     // d_state
#define DTR   16     // dt_rank
#define KCONV 4
#define NLAY  4
#define DBLC  48     // dt_rank + 2*d_state

__device__ __forceinline__ float sigmoidf_(float x) { return 1.f / (1.f + __expf(-x)); }

union FragU { v16bf v; v4u u[2]; };

// Load a 16x32 bf16 fragment (A or B^T, K-contiguous rows) per CDNA5 WMMA layout:
// lanes 0-15 hold K {0..7, 16..23}, lanes 16-31 hold K {8..15, 24..31}.
__device__ __forceinline__ v16bf load_kfrag(const bf16* __restrict__ p, int ld,
                                            int rbase, int k0, int lane) {
  int r   = lane & 15;
  int sel = (lane >> 4) << 3;               // 0 or 8
  const bf16* q = p + (size_t)(rbase + r) * ld + k0 + sel;
  FragU f;
  f.u[0] = *(const v4u*)(q);                // K = k0+sel .. +7
  f.u[1] = *(const v4u*)(q + 16);           // K = k0+16+sel .. +7
  return f.v;
}

// C(M,N) = A(M,K) @ W(N,K)^T, bf16 in / f32 accum.
// Each wave computes a 16 x (16*NSUB) strip: one A-fragment is reused across
// NSUB B-fragments -> NSUB WMMAs per (2 + 2*NSUB) b128 loads.
// K multiple of 32, M multiple of 16, N multiple of 16*NSUB.
template <int NSUB>
__global__ void gemm_bf16_wmma(const bf16* __restrict__ A, const bf16* __restrict__ W,
                               float* __restrict__ C, int M, int N, int K) {
  int lane = threadIdx.x & 31;
  int wave = threadIdx.x >> 5;
  int groupsN = N / (16 * NSUB);
  int grp = blockIdx.x * 4 + wave;
  if (grp >= (M >> 4) * groupsN) return;    // wave-uniform: EXEC all-1s for WMMA
  int tm  = (grp / groupsN) << 4;
  int tn0 = (grp % groupsN) * (16 * NSUB);

  v8f acc[NSUB];
#pragma unroll
  for (int j = 0; j < NSUB; ++j) acc[j] = (v8f){0.f, 0.f, 0.f, 0.f, 0.f, 0.f, 0.f, 0.f};

  for (int k0 = 0; k0 < K; k0 += 32) {
    v16bf a = load_kfrag(A, K, tm, k0, lane);
#pragma unroll
    for (int j = 0; j < NSUB; ++j) {
      v16bf b = load_kfrag(W, K, tn0 + j * 16, k0, lane);
      acc[j] = __builtin_amdgcn_wmma_f32_16x16x32_bf16(false, a, false, b,
                                                       (short)0, acc[j], false, false);
    }
  }

  int row = tm + ((lane >> 4) << 3);        // VGPR i -> M = i (lo lanes) / i+8 (hi lanes)
#pragma unroll
  for (int j = 0; j < NSUB; ++j) {
    int col = tn0 + j * 16 + (lane & 15);
#pragma unroll
    for (int i = 0; i < 8; ++i)
      C[(size_t)(row + i) * N + col] = acc[j][i];
  }
}

// f32 -> bf16 staging with optional zero-padding of columns and optional L-reversal.
__global__ void cvt2d_kernel(const float* __restrict__ src, bf16* __restrict__ dst,
                             int rows, int srcStride, int srcCols, int dstCols, int reverseL) {
  int idx = blockIdx.x * blockDim.x + threadIdx.x;
  int tot = rows * dstCols;
  if (idx >= tot) return;
  int r = idx / dstCols, c = idx - r * dstCols;
  int sr = r;
  if (reverseL) { int b = r / LSEQ, l = r % LSEQ; sr = b * LSEQ + (LSEQ - 1 - l); }
  float v = (c < srcCols) ? src[(size_t)sr * srcStride + c] : 0.f;
  dst[idx] = (bf16)v;
}

// Causal depthwise conv (K=4) + bias + SiLU on the xc half of xz.
__global__ void conv_silu_kernel(const float* __restrict__ xz, const float* __restrict__ cw,
                                 const float* __restrict__ cb, float* __restrict__ xconv) {
  int idx = blockIdx.x * blockDim.x + threadIdx.x;
  if (idx >= TOK * DI) return;
  int d = idx & (DI - 1);
  int tok = idx >> 9;
  int l = tok & (LSEQ - 1);
  float acc = cb[d];
#pragma unroll
  for (int k = 0; k < KCONV; ++k) {
    int ls = l + k - (KCONV - 1);
    if (ls >= 0)
      acc += xz[(size_t)(tok + k - (KCONV - 1)) * (2 * DI) + d] * cw[d * KCONV + k];
  }
  xconv[idx] = acc * sigmoidf_(acc);
}

__global__ void softplus_kernel(float* __restrict__ dt, const float* __restrict__ bdt) {
  int idx = blockIdx.x * blockDim.x + threadIdx.x;
  if (idx >= TOK * DI) return;
  int d = idx & (DI - 1);
  float v = dt[idx] + bdt[d];
  dt[idx] = (v > 20.f) ? v : log1pf(__expf(v));
}

// Selective scan: one thread per (batch, channel); 16 states in registers; 1024 steps.
__global__ void scan_kernel(const float* __restrict__ dt, const float* __restrict__ dbl,
                            const float* __restrict__ xconv, const float* __restrict__ A_log,
                            float* __restrict__ ys) {
  int gid = blockIdx.x * blockDim.x + threadIdx.x;
  if (gid >= NB * DI) return;
  int b = gid >> 9;
  int d = gid & (DI - 1);
  float Ar[DS], h[DS];
#pragma unroll
  for (int s = 0; s < DS; ++s) { Ar[s] = -__expf(A_log[d * DS + s]); h[s] = 0.f; }
  for (int t = 0; t < LSEQ; ++t) {
    int tok = b * LSEQ + t;
    float dtv = dt[(size_t)tok * DI + d];
    float xv  = xconv[(size_t)tok * DI + d];
    const float* bc = dbl + (size_t)tok * DBLC;
    float y = 0.f;
#pragma unroll
    for (int s = 0; s < DS; ++s) {
      float da = __expf(dtv * Ar[s]);
      float hs = fmaf(da, h[s], dtv * bc[DTR + s] * xv);
      h[s] = hs;
      y = fmaf(hs, bc[DTR + DS + s], y);
    }
    ys[(size_t)tok * DI + d] = y;
  }
}

// y = (ys + xconv*D) * silu(z), written straight into the bf16 GEMM staging buffer.
__global__ void gate_stage_kernel(const float* __restrict__ ys, const float* __restrict__ xconv,
                                  const float* __restrict__ Dp, const float* __restrict__ xz,
                                  bf16* __restrict__ abf) {
  int idx = blockIdx.x * blockDim.x + threadIdx.x;
  if (idx >= TOK * DI) return;
  int d = idx & (DI - 1);
  int tok = idx >> 9;
  float y = ys[idx] + xconv[idx] * Dp[d];
  float z = xz[(size_t)tok * (2 * DI) + DI + d];
  abf[idx] = (bf16)(y * (z * sigmoidf_(z)));
}

// Write a (TOK, DM) result into out (TOK, 2*DM) at column offset, optionally reversing L.
__global__ void concat_kernel(const float* __restrict__ res, float* __restrict__ out,
                              int off, int reverseL) {
  int idx = blockIdx.x * blockDim.x + threadIdx.x;
  if (idx >= TOK * DM) return;
  int tok = idx >> 8;
  int m = idx & (DM - 1);
  int b = tok >> 10, l = tok & (LSEQ - 1);
  int ol = reverseL ? (LSEQ - 1 - l) : l;
  out[(size_t)(b * LSEQ + ol) * (2 * DM) + off + m] = res[idx];
}

extern "C" void kernel_launch(void* const* d_in, const int* in_sizes, int n_in,
                              void* d_out, int out_size, void* d_ws, size_t ws_size,
                              hipStream_t stream) {
  const float* x        = (const float*)d_in[0];
  const float* in_proj  = (const float*)d_in[1];
  const float* conv_w   = (const float*)d_in[2];
  const float* conv_b   = (const float*)d_in[3];
  const float* x_proj   = (const float*)d_in[4];
  const float* dt_proj  = (const float*)d_in[5];
  const float* dt_bias  = (const float*)d_in[6];
  const float* A_log    = (const float*)d_in[7];
  const float* Dp       = (const float*)d_in[8];
  const float* out_proj = (const float*)d_in[9];
  float* out = (float*)d_out;
  (void)in_sizes; (void)n_in; (void)out_size; (void)ws_size;

  char* ws = (char*)d_ws;
  size_t off = 0;
  auto walloc = [&](size_t bytes) -> char* {
    char* p = ws + off;
    off += (bytes + 255) & ~(size_t)255;
    return p;
  };
  bf16* w_in_bf  = (bf16*)walloc(sizeof(bf16) * NLAY * 2 * DI * DM);
  bf16* w_x_bf   = (bf16*)walloc(sizeof(bf16) * NLAY * DBLC * DI);
  bf16* w_dt_bf  = (bf16*)walloc(sizeof(bf16) * NLAY * DI * 32);   // K padded 16->32
  bf16* w_out_bf = (bf16*)walloc(sizeof(bf16) * NLAY * DM * DI);
  bf16* a_bf     = (bf16*)walloc(sizeof(bf16) * TOK * DI);         // shared activation staging
  float* xz      = (float*)walloc(sizeof(float) * TOK * 2 * DI);
  float* xconv   = (float*)walloc(sizeof(float) * TOK * DI);
  float* dbl     = (float*)walloc(sizeof(float) * TOK * DBLC);
  float* dtb     = (float*)walloc(sizeof(float) * TOK * DI);
  float* ysb     = (float*)walloc(sizeof(float) * TOK * DI);
  float* xnext   = (float*)walloc(sizeof(float) * TOK * DM);

  const int TB = 256;
  auto cvt = [&](const float* src, bf16* dst, int rows, int sstride, int scols, int dcols, int rev) {
    int tot = rows * dcols;
    cvt2d_kernel<<<(tot + TB - 1) / TB, TB, 0, stream>>>(src, dst, rows, sstride, scols, dcols, rev);
  };
  auto gemm = [&](const bf16* A, const bf16* W, float* C, int M, int N, int K) {
    if ((N & 63) == 0) {
      int groups = (M / 16) * (N / 64);
      gemm_bf16_wmma<4><<<(groups + 3) / 4, 128, 0, stream>>>(A, W, C, M, N, K);
    } else {                                  // N = 48 (x_proj): 16x48 strip per wave
      int groups = (M / 16) * (N / 48);
      gemm_bf16_wmma<3><<<(groups + 3) / 4, 128, 0, stream>>>(A, W, C, M, N, K);
    }
  };

  // Stage all weights to bf16 once per launch (dt_proj K-padded to 32).
  for (int li = 0; li < NLAY; ++li) {
    cvt(in_proj  + (size_t)li * 2 * DI * DM, w_in_bf  + (size_t)li * 2 * DI * DM, 2 * DI, DM,  DM,  DM, 0);
    cvt(x_proj   + (size_t)li * DBLC * DI,   w_x_bf   + (size_t)li * DBLC * DI,   DBLC,   DI,  DI,  DI, 0);
    cvt(dt_proj  + (size_t)li * DI * DTR,    w_dt_bf  + (size_t)li * DI * 32,     DI,     DTR, DTR, 32, 0);
    cvt(out_proj + (size_t)li * DM * DI,     w_out_bf + (size_t)li * DM * DI,     DM,     DI,  DI,  DI, 0);
  }

  auto run_layer = [&](int li, const float* xin, int revIn) {
    cvt(xin, a_bf, TOK, DM, DM, DM, revIn);                                    // stage input
    gemm(a_bf, w_in_bf + (size_t)li * 2 * DI * DM, xz, TOK, 2 * DI, DM);       // in_proj
    conv_silu_kernel<<<(TOK * DI + TB - 1) / TB, TB, 0, stream>>>(
        xz, conv_w + (size_t)li * DI * KCONV, conv_b + (size_t)li * DI, xconv);
    cvt(xconv, a_bf, TOK, DI, DI, DI, 0);
    gemm(a_bf, w_x_bf + (size_t)li * DBLC * DI, dbl, TOK, DBLC, DI);           // x_proj
    cvt(dbl, a_bf, TOK, DBLC, DTR, 32, 0);                                     // dt_raw, K-pad
    gemm(a_bf, w_dt_bf + (size_t)li * DI * 32, dtb, TOK, DI, 32);              // dt_proj
    softplus_kernel<<<(TOK * DI + TB - 1) / TB, TB, 0, stream>>>(dtb, dt_bias + (size_t)li * DI);
    scan_kernel<<<(NB * DI + 127) / 128, 128, 0, stream>>>(
        dtb, dbl, xconv, A_log + (size_t)li * DI * DS, ysb);
    gate_stage_kernel<<<(TOK * DI + TB - 1) / TB, TB, 0, stream>>>(
        ysb, xconv, Dp + (size_t)li * DI, xz, a_bf);
    gemm(a_bf, w_out_bf + (size_t)li * DM * DI, xnext, TOK, DM, DI);           // out_proj
  };

  // Forward chain (layers 0,1)
  run_layer(0, x, 0);
  run_layer(1, xnext, 0);
  concat_kernel<<<(TOK * DM + TB - 1) / TB, TB, 0, stream>>>(xnext, out, 0, 0);

  // Backward chain (layers 2,3) on L-reversed input; un-reverse on concat.
  run_layer(2, x, 1);
  run_layer(3, xnext, 0);
  concat_kernel<<<(TOK * DM + TB - 1) / TB, TB, 0, stream>>>(xnext, out, DM, 1);
}
